// OperatorBlocks_53953379172720
// MI455X (gfx1250) — matrix operand order repaired and checked by reading, not compiled, not run on hardware
//
#include <hip/hip_runtime.h>
#include <hip/hip_bf16.h>
#include <math.h>

// ---------------- problem constants (match reference setup) ----------------
#define BB   8
#define NNODE 2048
#define CCH  256
#define EE   32768
#define LLAY 3
#define KCH  3
#define BNC  (BB * NNODE * CCH)   // 4,194,304 elements
#define MM   (BB * NNODE)         // 16384 rows for GEMMs
#define NWMAT (LLAY * KCH + LLAY) // 12 weight matrices of 256x256
#define WELEMS (NWMAT * CCH * CCH)

typedef __attribute__((ext_vector_type(16))) __bf16 bf16x16;
typedef __attribute__((ext_vector_type(8)))  float  f32x8;
typedef __attribute__((ext_vector_type(4)))  float  f32x4;

// ================= weight prep: transpose + split to bf16 hi/lo ============
// wt_hi/wt_lo layout: [mat][n][k]  (i.e. W^T, row-major over n, contiguous k)
__global__ void prep_weights(const float* __restrict__ cheb_w,
                             const float* __restrict__ mlp_w,
                             __bf16* __restrict__ wt_hi,
                             __bf16* __restrict__ wt_lo) {
  for (int idx = blockIdx.x * blockDim.x + threadIdx.x; idx < WELEMS;
       idx += gridDim.x * blockDim.x) {
    int mat = idx >> 16;          // /65536
    int rem = idx & 65535;
    int n = rem >> 8;             // output channel
    int c = rem & 255;            // input channel (k)
    float v = (mat < LLAY * KCH)
                  ? cheb_w[(mat << 16) + (c << 8) + n]
                  : mlp_w[((mat - LLAY * KCH) << 16) + (c << 8) + n];
    __bf16 hv = (__bf16)v;
    wt_hi[idx] = hv;
    wt_lo[idx] = (__bf16)(v - (float)hv);
  }
}

// ================= edge preprocessing ======================================
__global__ void edge_deg(const float* __restrict__ w, const int* __restrict__ row,
                         const int* __restrict__ col, float* __restrict__ deg) {
  int e = blockIdx.x * blockDim.x + threadIdx.x;
  if (e >= EE) return;
  int r = row[e], c = col[e];
  float we = w[e];
  if (r != c)
    __hip_atomic_fetch_add(&deg[r], we, __ATOMIC_RELAXED, __HIP_MEMORY_SCOPE_AGENT);
}

__global__ void edge_norm(const float* __restrict__ w, const int* __restrict__ row,
                          const int* __restrict__ col, const float* __restrict__ deg,
                          float* __restrict__ nrm) {
  int e = blockIdx.x * blockDim.x + threadIdx.x;
  if (e >= EE) return;
  int r = row[e], c = col[e];
  float we = w[e];
  float dr = deg[r], dc = deg[c];
  float ir = (dr > 0.f) ? rsqrtf(fmaxf(dr, 1e-12f)) : 0.f;
  float ic = (dc > 0.f) ? rsqrtf(fmaxf(dc, 1e-12f)) : 0.f;
  nrm[e] = (r == c) ? 0.f : -(ir * we * ic);
}

// ================= global LayerNorm (mean/var over whole tensor) ===========
__global__ void reduce_partial(const float* __restrict__ x, float* __restrict__ part) {
  __shared__ float ss[256], sq[256];
  int tid = threadIdx.x;
  float s = 0.f, q = 0.f;
  for (int i = blockIdx.x * 256 + tid; i < BNC; i += gridDim.x * 256) {
    float v = x[i];
    s += v;
    q += v * v;
  }
  ss[tid] = s; sq[tid] = q;
  __syncthreads();
  for (int o = 128; o > 0; o >>= 1) {
    if (tid < o) { ss[tid] += ss[tid + o]; sq[tid] += sq[tid + o]; }
    __syncthreads();
  }
  if (tid == 0) { part[2 * blockIdx.x] = ss[0]; part[2 * blockIdx.x + 1] = sq[0]; }
}

__global__ void reduce_final(const float* __restrict__ part, float* __restrict__ stats,
                             int nblk) {
  __shared__ float ss[256], sq[256];
  int tid = threadIdx.x;
  float s = 0.f, q = 0.f;
  for (int i = tid; i < nblk; i += 256) { s += part[2 * i]; q += part[2 * i + 1]; }
  ss[tid] = s; sq[tid] = q;
  __syncthreads();
  for (int o = 128; o > 0; o >>= 1) {
    if (tid < o) { ss[tid] += ss[tid + o]; sq[tid] += sq[tid + o]; }
    __syncthreads();
  }
  if (tid == 0) {
    const float inv = 1.f / (float)BNC;
    float m = ss[0] * inv;
    float var = sq[0] * inv - m * m;
    stats[0] = m;
    stats[1] = rsqrtf(var + 1e-5f);
  }
}

__global__ void ln_apply(const float* __restrict__ h, const float* __restrict__ stats,
                         float* __restrict__ t0) {
  float m = stats[0], r = stats[1];
  for (int i = blockIdx.x * blockDim.x + threadIdx.x; i < BNC;
       i += gridDim.x * blockDim.x)
    t0[i] = (h[i] - m) * r;
}

// ================= Cheb propagation (scatter) ==============================
// dst[:, col[e], :] += nrm[e] * src[:, row[e], :]   — hardware f32 atomics
__global__ void prop_kernel(const float* __restrict__ src, float* __restrict__ dst,
                            const float* __restrict__ nrm,
                            const int* __restrict__ row, const int* __restrict__ col) {
  int e = blockIdx.x;
  float ne = nrm[e];
  int r = row[e], c = col[e];
  int ch = threadIdx.x;
  const float* sp = src + (size_t)r * CCH + ch;
  float* dp = dst + (size_t)c * CCH + ch;
#pragma unroll
  for (int b = 0; b < BB; ++b)
    __hip_atomic_fetch_add(dp + (size_t)b * NNODE * CCH, ne * sp[(size_t)b * NNODE * CCH],
                           __ATOMIC_RELAXED, __HIP_MEMORY_SCOPE_AGENT);
}

// Tx2 = 2*P - Tx0  (in place into t0)
__global__ void cheb2_combine(const float* __restrict__ p, float* __restrict__ t0) {
  for (int i = blockIdx.x * blockDim.x + threadIdx.x; i < BNC;
       i += gridDim.x * blockDim.x)
    t0[i] = 2.f * p[i] - t0[i];
}

// h += gemm_out + cheb_b[c]
__global__ void cheb_residual(float* __restrict__ h, const float* __restrict__ o,
                              const float* __restrict__ bias) {
  for (int i = blockIdx.x * blockDim.x + threadIdx.x; i < BNC;
       i += gridDim.x * blockDim.x)
    h[i] += o[i] + bias[i & (CCH - 1)];
}

// h += gelu_exact(gemm_out + mlp_b[c])
__global__ void mlp_residual(float* __restrict__ h, const float* __restrict__ o,
                             const float* __restrict__ bias) {
  for (int i = blockIdx.x * blockDim.x + threadIdx.x; i < BNC;
       i += gridDim.x * blockDim.x) {
    float x = o[i] + bias[i & (CCH - 1)];
    h[i] += 0.5f * x * (1.f + erff(x * 0.70710678118654752f));
  }
}

__global__ void final_mean(const float* __restrict__ h, float* __restrict__ out) {
  int b = blockIdx.x, c = threadIdx.x;
  float s = 0.f;
  for (int n = 0; n < NNODE; ++n) s += h[((size_t)b * NNODE + n) * CCH + c];
  out[b * CCH + c] = s * (1.f / (float)NNODE);
}

// ================= WMMA GEMM: C[M,256] (+)= A[M,256] @ W[256,256] ==========
// A is f32 row-major; weights are pre-split bf16 W^T (hi/lo), [n][k] row-major.
// Split-bf16 trick: A = Ahi + Alo, acc += Ahi*Bhi + Alo*Bhi + Ahi*Blo (f32 acc)
// gives ~f32-quality results on the bf16 matrix pipe.
#define GEMM_WAVES 4
__global__ void __launch_bounds__(32 * GEMM_WAVES)
gemm_wmma_bf16split(const float* __restrict__ A, const __bf16* __restrict__ BThi,
                    const __bf16* __restrict__ BTlo, float* __restrict__ C,
                    int accumulate) {
  const int lane = threadIdx.x & 31;
  const int wave = threadIdx.x >> 5;
  const int half = lane >> 4;
  const int l16  = lane & 15;
  const int tileM = (blockIdx.x * GEMM_WAVES + wave) * 16;
  const int tileN = blockIdx.y * 64;

  f32x8 acc[4];
  if (accumulate) {
#pragma unroll
    for (int j = 0; j < 4; ++j) {
      const float* cp = C + (size_t)(tileM + 8 * half) * CCH + tileN + j * 16 + l16;
#pragma unroll
      for (int g = 0; g < 8; ++g) acc[j][g] = cp[(size_t)g * CCH];
    }
  } else {
#pragma unroll
    for (int j = 0; j < 4; ++j)
#pragma unroll
      for (int g = 0; g < 8; ++g) acc[j][g] = 0.f;
  }

  const float* arow = A + (size_t)(tileM + l16) * CCH;

#pragma unroll
  for (int kk = 0; kk < CCH; kk += 32) {
    // ---- A fragment (contiguous k order, same as B: lanes0-15 K={0-15}+kk,
    //      lanes16-31 K={16-31}+kk), converted f32 -> bf16 hi/lo
    const int ka = kk + 16 * half;
    f32x4 a0 = *(const f32x4*)(arow + ka);
    f32x4 a1 = *(const f32x4*)(arow + ka + 4);
    f32x4 a2 = *(const f32x4*)(arow + ka + 8);
    f32x4 a3 = *(const f32x4*)(arow + ka + 12);
    bf16x16 afh, afl;
#pragma unroll
    for (int i = 0; i < 4; ++i) {
      float v;
      v = a0[i]; afh[i]      = (__bf16)v; afl[i]      = (__bf16)(v - (float)afh[i]);
      v = a1[i]; afh[4 + i]  = (__bf16)v; afl[4 + i]  = (__bf16)(v - (float)afh[4 + i]);
      v = a2[i]; afh[8 + i]  = (__bf16)v; afl[8 + i]  = (__bf16)(v - (float)afh[8 + i]);
      v = a3[i]; afh[12 + i] = (__bf16)v; afl[12 + i] = (__bf16)(v - (float)afh[12 + i]);
    }
    // ---- B fragments: lane holds column n=l16, contiguous K = kk + 16*half
    const int kb = kk + 16 * half;
#pragma unroll
    for (int j = 0; j < 4; ++j) {
      const size_t boff = (size_t)(tileN + j * 16 + l16) * CCH + kb;
      bf16x16 bh = *(const bf16x16*)(BThi + boff);
      bf16x16 bl = *(const bf16x16*)(BTlo + boff);
      acc[j] = __builtin_amdgcn_wmma_f32_16x16x32_bf16(false, afh, false, bh,
                                                       (short)0, acc[j], false, false);
      acc[j] = __builtin_amdgcn_wmma_f32_16x16x32_bf16(false, afl, false, bh,
                                                       (short)0, acc[j], false, false);
      acc[j] = __builtin_amdgcn_wmma_f32_16x16x32_bf16(false, afh, false, bl,
                                                       (short)0, acc[j], false, false);
    }
  }

#pragma unroll
  for (int j = 0; j < 4; ++j) {
    float* cp = C + (size_t)(tileM + 8 * half) * CCH + tileN + j * 16 + l16;
#pragma unroll
    for (int g = 0; g < 8; ++g) cp[(size_t)g * CCH] = acc[j][g];
  }
}

// ================= driver ==================================================
extern "C" void kernel_launch(void* const* d_in, const int* in_sizes, int n_in,
                              void* d_out, int out_size, void* d_ws, size_t ws_size,
                              hipStream_t stream) {
  const float* node_feature = (const float*)d_in[0];
  const float* edge_weight  = (const float*)d_in[1];
  const float* cheb_w       = (const float*)d_in[2];
  const float* cheb_b       = (const float*)d_in[3];
  const float* mlp_w        = (const float*)d_in[4];
  const float* mlp_b        = (const float*)d_in[5];
  const int*   edge_index   = (const int*)d_in[6];
  float* out = (float*)d_out;

  float* ws = (float*)d_ws;
  float* h   = ws;               // [B,N,C] evolving activations
  float* t0  = h  + BNC;         // Tx0 / LN output
  float* t1  = t0 + BNC;         // Tx1
  float* pp  = t1 + BNC;         // prop scratch
  float* oo  = pp + BNC;         // GEMM output accumulator
  __bf16* wt_hi = (__bf16*)(oo + BNC);
  __bf16* wt_lo = wt_hi + WELEMS;
  float* deg   = (float*)(wt_lo + WELEMS);
  float* nrm   = deg + NNODE;
  float* part  = nrm + EE;       // 2*1024 partials
  float* stats = part + 2 * 1024;

  hipMemcpyAsync(h, node_feature, (size_t)BNC * sizeof(float),
                 hipMemcpyDeviceToDevice, stream);
  prep_weights<<<256, 256, 0, stream>>>(cheb_w, mlp_w, wt_hi, wt_lo);

  dim3 gemm_grid(MM / (16 * GEMM_WAVES), CCH / 64);  // (256, 4)
  const int gemm_block = 32 * GEMM_WAVES;

  for (int l = 0; l < LLAY; ++l) {
    const int* row = edge_index + (size_t)l * 2 * EE;
    const int* col = row + EE;
    const float* ew = edge_weight + (size_t)l * EE;
    const __bf16* Wc_hi = wt_hi + (size_t)(l * KCH) * CCH * CCH;
    const __bf16* Wc_lo = wt_lo + (size_t)(l * KCH) * CCH * CCH;
    const __bf16* Wm_hi = wt_hi + (size_t)(LLAY * KCH + l) * CCH * CCH;
    const __bf16* Wm_lo = wt_lo + (size_t)(LLAY * KCH + l) * CCH * CCH;

    // --- edge normalization for this layer
    hipMemsetAsync(deg, 0, NNODE * sizeof(float), stream);
    edge_deg<<<EE / 256, 256, 0, stream>>>(ew, row, col, deg);
    edge_norm<<<EE / 256, 256, 0, stream>>>(ew, row, col, deg, nrm);

    // --- LN #1 -> t0
    reduce_partial<<<1024, 256, 0, stream>>>(h, part);
    reduce_final<<<1, 256, 0, stream>>>(part, stats, 1024);
    ln_apply<<<1024, 256, 0, stream>>>(h, stats, t0);

    // --- Cheb: out = Tx0 @ W0
    gemm_wmma_bf16split<<<gemm_grid, gemm_block, 0, stream>>>(t0, Wc_hi, Wc_lo, oo, 0);

    // --- Tx1 = prop(Tx0);  out += Tx1 @ W1
    hipMemsetAsync(t1, 0, (size_t)BNC * sizeof(float), stream);
    prop_kernel<<<EE, CCH, 0, stream>>>(t0, t1, nrm, row, col);
    gemm_wmma_bf16split<<<gemm_grid, gemm_block, 0, stream>>>(
        t1, Wc_hi + CCH * CCH, Wc_lo + CCH * CCH, oo, 1);

    // --- Tx2 = 2*prop(Tx1) - Tx0 (into t0);  out += Tx2 @ W2
    hipMemsetAsync(pp, 0, (size_t)BNC * sizeof(float), stream);
    prop_kernel<<<EE, CCH, 0, stream>>>(t1, pp, nrm, row, col);
    cheb2_combine<<<1024, 256, 0, stream>>>(pp, t0);
    gemm_wmma_bf16split<<<gemm_grid, gemm_block, 0, stream>>>(
        t0, Wc_hi + 2 * CCH * CCH, Wc_lo + 2 * CCH * CCH, oo, 1);

    // --- residual: h += out + cheb_b
    cheb_residual<<<1024, 256, 0, stream>>>(h, oo, cheb_b + l * CCH);

    // --- LN #2 -> t0 ; MLP GEMM ; gelu + residual
    reduce_partial<<<1024, 256, 0, stream>>>(h, part);
    reduce_final<<<1, 256, 0, stream>>>(part, stats, 1024);
    ln_apply<<<1024, 256, 0, stream>>>(h, stats, t0);
    gemm_wmma_bf16split<<<gemm_grid, gemm_block, 0, stream>>>(t0, Wm_hi, Wm_lo, oo, 0);
    mlp_residual<<<1024, 256, 0, stream>>>(h, oo, mlp_b + l * CCH);
  }

  final_mean<<<BB, CCH, 0, stream>>>(h, out);

  (void)in_sizes; (void)n_in; (void)out_size; (void)ws_size;
}
